// DualAttention_15753940042189
// MI455X (gfx1250) — matrix operand order repaired and checked by reading, not compiled, not run on hardware
//
#include <hip/hip_runtime.h>
#include <stdint.h>

#define DEVFN __device__ __forceinline__

typedef __attribute__((ext_vector_type(16))) __bf16 v16bf;
typedef __attribute__((ext_vector_type(8)))  __bf16 v8bf;
typedef __attribute__((ext_vector_type(8)))  float  v8f;
typedef __attribute__((ext_vector_type(4)))  float  f32x4;

constexpr int B_  = 8;
constexpr int S1_ = 2048;
constexpr int S2_ = 2048;
constexpr int H_  = 1024;

#define WMMA_BF16(A_, Bm_, C_) \
  __builtin_amdgcn_wmma_f32_16x16x32_bf16(false, (A_), false, (Bm_), (short)0, (C_), false, false)

// ---------------- fragment helpers (layouts per CDNA5 ISA 7.12.2) ----------------

DEVFN v16bf cat16(v8bf lo, v8bf hi) {
  v16bf r;
#pragma unroll
  for (int e = 0; e < 8; ++e) { r[e] = lo[e]; r[8 + e] = hi[e]; }
  return r;
}

// A fragment 16(M)x32(K), bf16 row-major source; A = &src[m0*lda + k0]
DEVFN v16bf ldA_bf16(const __bf16* A, int lda, int lane) {
  int r = lane & 15, s = (lane >> 4) * 8;
  const __bf16* p = A + (size_t)r * lda + s;
  return cat16(*(const v8bf*)p, *(const v8bf*)(p + 16));
}

// A fragment from f32 row-major, convert to bf16 on the fly
DEVFN v16bf ldA_f32(const float* A, int lda, int lane) {
  int r = lane & 15, s = (lane >> 4) * 8;
  const float* p = A + (size_t)r * lda + s;
  f32x4 f0 = *(const f32x4*)p,        f1 = *(const f32x4*)(p + 4);
  f32x4 f2 = *(const f32x4*)(p + 16), f3 = *(const f32x4*)(p + 20);
  v16bf r16;
#pragma unroll
  for (int e = 0; e < 4; ++e) {
    r16[e] = (__bf16)f0[e];      r16[4 + e]  = (__bf16)f1[e];
    r16[8 + e] = (__bf16)f2[e];  r16[12 + e] = (__bf16)f3[e];
  }
  return r16;
}

// A fragment = exp(w - rowmax[r]) from f32 row-major (flash-style P recompute)
DEVFN v16bf ldA_f32exp(const float* A, int lda, const float* rowmax, int lane) {
  int r = lane & 15, s = (lane >> 4) * 8;
  const float* p = A + (size_t)r * lda + s;
  float m = rowmax[r];
  f32x4 f0 = *(const f32x4*)p,        f1 = *(const f32x4*)(p + 4);
  f32x4 f2 = *(const f32x4*)(p + 16), f3 = *(const f32x4*)(p + 20);
  v16bf r16;
#pragma unroll
  for (int e = 0; e < 4; ++e) {
    r16[e]      = (__bf16)__expf(f0[e] - m);
    r16[4 + e]  = (__bf16)__expf(f1[e] - m);
    r16[8 + e]  = (__bf16)__expf(f2[e] - m);
    r16[12 + e] = (__bf16)__expf(f3[e] - m);
  }
  return r16;
}

// B fragment 32(K)x16(N): lane owns column n, contiguous K ("BT" layout: BT[n*ldb + k])
DEVFN v16bf ldB_T_bf16(const __bf16* BT, int ldb, int lane) {
  int n = lane & 15, s = (lane >> 4) * 8;
  const __bf16* p = BT + (size_t)n * ldb + s;
  return cat16(*(const v8bf*)p, *(const v8bf*)(p + 16));
}

// B fragment whose columns are contiguous f32 rows of the source (convert on load)
DEVFN v16bf ldB_T_f32(const float* BT, int ldb, int lane) {
  int n = lane & 15, s = (lane >> 4) * 8;
  const float* p = BT + (size_t)n * ldb + s;
  f32x4 f0 = *(const f32x4*)p,        f1 = *(const f32x4*)(p + 4);
  f32x4 f2 = *(const f32x4*)(p + 16), f3 = *(const f32x4*)(p + 20);
  v16bf r16;
#pragma unroll
  for (int e = 0; e < 4; ++e) {
    r16[e] = (__bf16)f0[e];      r16[4 + e]  = (__bf16)f1[e];
    r16[8 + e] = (__bf16)f2[e];  r16[12 + e] = (__bf16)f3[e];
  }
  return r16;
}

// D stores: lane covers column n = lane&15, rows m = e + 8*(lane>>4)
DEVFN void stD_bf16(__bf16* C, int ldc, v8f d, int lane) {
  int n = lane & 15, hi = lane >> 4;
#pragma unroll
  for (int e = 0; e < 8; ++e)
    C[(size_t)(e + hi * 8) * ldc + n] = (__bf16)d[e];
}

DEVFN void stD_f32_bias(float* C, int ldc, float bias, v8f d, int lane) {
  int n = lane & 15, hi = lane >> 4;
#pragma unroll
  for (int e = 0; e < 8; ++e)
    C[(size_t)(e + hi * 8) * ldc + n] = d[e] + bias;
}

DEVFN void stD_f32_scale(float* C, int ldc, const float* sc, v8f d, int lane) {
  int n = lane & 15, hi = lane >> 4;
#pragma unroll
  for (int e = 0; e < 8; ++e) {
    int m = e + hi * 8;
    C[(size_t)m * ldc + n] = d[e] * sc[m];
  }
}

// fragment bundle: one A tile + four B tiles (16M x 64N per wave)
struct Frag { v16bf a, b0, b1, b2, b3; };

DEVFN void mma4(const Frag& f, v8f& c0, v8f& c1, v8f& c2, v8f& c3) {
  c0 = WMMA_BF16(f.a, f.b0, c0);
  c1 = WMMA_BF16(f.a, f.b1, c1);
  c2 = WMMA_BF16(f.a, f.b2, c2);
  c3 = WMMA_BF16(f.a, f.b3, c3);
}

// ---------------- kernel 0: tiled transpose + f32->bf16 convert ----------------

__global__ __launch_bounds__(256) void k0_transpose_cvt(const float* __restrict__ in,
                                                        __bf16* __restrict__ out,
                                                        int rows, int cols) {
  __shared__ float tile[32][33];
  const float* src = in + (size_t)blockIdx.z * rows * cols;
  __bf16* dst = out + (size_t)blockIdx.z * rows * cols;
  int c0 = blockIdx.x * 32, r0 = blockIdx.y * 32;
  int tx = threadIdx.x & 31, ty = threadIdx.x >> 5;
#pragma unroll
  for (int i = 0; i < 32; i += 8)
    tile[ty + i][tx] = src[(size_t)(r0 + ty + i) * cols + c0 + tx];
  __syncthreads();
#pragma unroll
  for (int i = 0; i < 32; i += 8)
    dst[(size_t)(c0 + ty + i) * rows + r0 + tx] = (__bf16)tile[tx][ty + i];
}

// ------ kernel 1: h1w(bf16) = hidden1 @ W  (ping-pong double-buffered K loop) ------

DEVFN Frag ld_k1(const float* A, const __bf16* WTn, int k0, int lane) {
  Frag f;
  f.a  = ldA_f32(A + k0, H_, lane);
  f.b0 = ldB_T_bf16(WTn + (size_t) 0 * H_ + k0, H_, lane);
  f.b1 = ldB_T_bf16(WTn + (size_t)16 * H_ + k0, H_, lane);
  f.b2 = ldB_T_bf16(WTn + (size_t)32 * H_ + k0, H_, lane);
  f.b3 = ldB_T_bf16(WTn + (size_t)48 * H_ + k0, H_, lane);
  return f;
}

__global__ __launch_bounds__(128) void k1_h1w(const float* __restrict__ h1,
                                              const __bf16* __restrict__ WT,
                                              __bf16* __restrict__ h1w) {
  int lane = threadIdx.x & 31, wave = threadIdx.x >> 5;
  int b = blockIdx.z;
  int n0 = blockIdx.x * 64;
  int m0 = (blockIdx.y * 4 + wave) * 16;
  const float* A = h1 + ((size_t)b * S1_ + m0) * H_;
  const __bf16* WTn = WT + (size_t)n0 * H_;
  v8f c0 = {}, c1 = {}, c2 = {}, c3 = {};

  Frag f0 = ld_k1(A, WTn, 0, lane), f1;
  for (int k = 32; k <= H_ - 64; k += 64) {
    __builtin_prefetch(A + k + 64, 0, 3);
    f1 = ld_k1(A, WTn, k, lane);
    mma4(f0, c0, c1, c2, c3);
    f0 = ld_k1(A, WTn, k + 32, lane);
    mma4(f1, c0, c1, c2, c3);
  }
  mma4(f0, c0, c1, c2, c3);

  __bf16* D = h1w + ((size_t)b * S1_ + m0) * H_ + n0;
  stD_bf16(D,      H_, c0, lane);
  stD_bf16(D + 16, H_, c1, lane);
  stD_bf16(D + 32, H_, c2, lane);
  stD_bf16(D + 48, H_, c3, lane);
}

// ------ kernel 2: weight(f32) = h1w @ hidden2^T + bias (ping-pong) ------

DEVFN Frag ld_k2(const __bf16* A, const float* Bn, int k0, int lane) {
  Frag f;
  f.a  = ldA_bf16(A + k0, H_, lane);
  f.b0 = ldB_T_f32(Bn + (size_t) 0 * H_ + k0, H_, lane);
  f.b1 = ldB_T_f32(Bn + (size_t)16 * H_ + k0, H_, lane);
  f.b2 = ldB_T_f32(Bn + (size_t)32 * H_ + k0, H_, lane);
  f.b3 = ldB_T_f32(Bn + (size_t)48 * H_ + k0, H_, lane);
  return f;
}

__global__ __launch_bounds__(128) void k2_weight(const __bf16* __restrict__ h1w,
                                                 const float* __restrict__ h2,
                                                 const float* __restrict__ bias,
                                                 float* __restrict__ wgt) {
  int lane = threadIdx.x & 31, wave = threadIdx.x >> 5;
  int b = blockIdx.z;
  int n0 = blockIdx.x * 64;
  int m0 = (blockIdx.y * 4 + wave) * 16;
  const __bf16* A = h1w + ((size_t)b * S1_ + m0) * H_;
  const float* Bn = h2 + ((size_t)b * S2_ + n0) * H_;
  v8f c0 = {}, c1 = {}, c2 = {}, c3 = {};

  Frag f0 = ld_k2(A, Bn, 0, lane), f1;
  for (int k = 32; k <= H_ - 64; k += 64) {
    __builtin_prefetch(A + k + 64, 0, 3);
    f1 = ld_k2(A, Bn, k, lane);
    mma4(f0, c0, c1, c2, c3);
    f0 = ld_k2(A, Bn, k + 32, lane);
    mma4(f1, c0, c1, c2, c3);
  }
  mma4(f0, c0, c1, c2, c3);

  float bv = bias[0];
  float* D = wgt + ((size_t)b * S1_ + m0) * S2_ + n0;
  stD_f32_bias(D,      S2_, bv, c0, lane);
  stD_f32_bias(D + 16, S2_, bv, c1, lane);
  stD_f32_bias(D + 32, S2_, bv, c2, lane);
  stD_f32_bias(D + 48, S2_, bv, c3, lane);
}

// ---------------- kernel 3: row softmax stats (axis=2) ----------------

__global__ __launch_bounds__(256) void k3_rowstats(const float* __restrict__ wgt,
                                                   float* __restrict__ rmax,
                                                   float* __restrict__ rinv) {
  int row = blockIdx.x;  // b*S1 + i
  const float* p = wgt + (size_t)row * S2_;
  float m = -__builtin_inff(), s = 0.f;
  for (int j = threadIdx.x; j < S2_; j += 256) {
    float v = p[j];
    float nm = fmaxf(m, v);
    s = s * __expf(m - nm) + __expf(v - nm);
    m = nm;
  }
  __shared__ float sm[256], ss[256];
  sm[threadIdx.x] = m; ss[threadIdx.x] = s;
  __syncthreads();
  for (int off = 128; off >= 1; off >>= 1) {
    if ((int)threadIdx.x < off) {
      float m1 = sm[threadIdx.x], s1 = ss[threadIdx.x];
      float m2 = sm[threadIdx.x + off], s2 = ss[threadIdx.x + off];
      float nm = fmaxf(m1, m2);
      sm[threadIdx.x] = nm;
      ss[threadIdx.x] = s1 * __expf(m1 - nm) + s2 * __expf(m2 - nm);
    }
    __syncthreads();
  }
  if (threadIdx.x == 0) { rmax[row] = sm[0]; rinv[row] = 1.0f / ss[0]; }
}

// ---------------- kernel 4: column softmax stats (axis=1) ----------------

__global__ __launch_bounds__(256) void k4_colstats(const float* __restrict__ wgt,
                                                   float* __restrict__ cmax,
                                                   float* __restrict__ cinv) {
  int b = blockIdx.y;
  int j = blockIdx.x * 256 + threadIdx.x;
  const float* p = wgt + (size_t)b * S1_ * S2_ + j;
  float m = -__builtin_inff(), s = 0.f;
  for (int i = 0; i < S1_; ++i) {
    float v = p[(size_t)i * S2_];
    float nm = fmaxf(m, v);
    s = s * __expf(m - nm) + __expf(v - nm);
    m = nm;
  }
  cmax[(size_t)b * S2_ + j] = m;
  cinv[(size_t)b * S2_ + j] = 1.0f / s;
}

// ------- kernel 5: hidden_2 = softmax_row(weight) @ hidden2  (ping-pong) -------

DEVFN Frag ld_k5(const float* A, const __bf16* Bn, const float* mrow, int k0, int lane) {
  Frag f;
  f.a  = ldA_f32exp(A + k0, S2_, mrow, lane);
  f.b0 = ldB_T_bf16(Bn + (size_t) 0 * S2_ + k0, S2_, lane);
  f.b1 = ldB_T_bf16(Bn + (size_t)16 * S2_ + k0, S2_, lane);
  f.b2 = ldB_T_bf16(Bn + (size_t)32 * S2_ + k0, S2_, lane);
  f.b3 = ldB_T_bf16(Bn + (size_t)48 * S2_ + k0, S2_, lane);
  return f;
}

__global__ __launch_bounds__(128) void k5_hidden2(const float* __restrict__ wgt,
                                                  const __bf16* __restrict__ h2T,
                                                  const float* __restrict__ rmax,
                                                  const float* __restrict__ rinv,
                                                  float* __restrict__ out) {
  int lane = threadIdx.x & 31, wave = threadIdx.x >> 5;
  int b = blockIdx.z;
  int n0 = blockIdx.x * 64;
  int m0 = (blockIdx.y * 4 + wave) * 16;
  const float* A = wgt + ((size_t)b * S1_ + m0) * S2_;
  const float* mrow = rmax + (size_t)b * S1_ + m0;
  const __bf16* Bn = h2T + ((size_t)b * H_ + n0) * S2_;
  v8f c0 = {}, c1 = {}, c2 = {}, c3 = {};

  Frag f0 = ld_k5(A, Bn, mrow, 0, lane), f1;
  for (int k = 32; k <= S2_ - 64; k += 64) {
    __builtin_prefetch(A + k + 64, 0, 3);
    f1 = ld_k5(A, Bn, mrow, k, lane);
    mma4(f0, c0, c1, c2, c3);
    f0 = ld_k5(A, Bn, mrow, k + 32, lane);
    mma4(f1, c0, c1, c2, c3);
  }
  mma4(f0, c0, c1, c2, c3);

  const float* sc = rinv + (size_t)b * S1_ + m0;
  float* D = out + ((size_t)b * S1_ + m0) * H_ + n0;
  stD_f32_scale(D,      H_, sc, c0, lane);
  stD_f32_scale(D + 16, H_, sc, c1, lane);
  stD_f32_scale(D + 32, H_, sc, c2, lane);
  stD_f32_scale(D + 48, H_, sc, c3, lane);
}

// ------- kernel 6: hidden_1 = softmax_col(weight)^T @ hidden1  (B,S2,H) -------
// A-matrix (weight^T tile) staged through LDS with async global->LDS copies so the
// global reads stay coalesced; the transpose + exp happens on the LDS read side.

__global__ __launch_bounds__(128) void k6_hidden1(const float* __restrict__ wgt,
                                                  const __bf16* __restrict__ h1T,
                                                  const float* __restrict__ cmax,
                                                  const float* __restrict__ cinv,
                                                  float* __restrict__ out) {
  __shared__ float atile[32][68];   // 32 K-rows x 64 j-cols (+pad)
  int tid = threadIdx.x;
  int lane = tid & 31, wave = tid >> 5;
  int b = blockIdx.z;
  int n0 = blockIdx.x * 64;
  int mblk = blockIdx.y * 64;            // block's j-range
  int m0 = mblk + wave * 16;             // wave's j-range (rows of output)
  const float* Wb = wgt + (size_t)b * S1_ * S2_ + mblk;
  const __bf16* Bb = h1T + ((size_t)b * H_ + n0) * S1_;
  float mx = cmax[(size_t)b * S2_ + m0 + (lane & 15)];

  // staging coordinates: 128 threads x 4 iters x 16B = 32x64 f32 tile
  int sr = tid >> 4;            // 0..7
  int sc = (tid & 15) * 4;      // 0..60

  v8f c0 = {}, c1 = {}, c2 = {}, c3 = {};
  for (int k0 = 0; k0 < S1_; k0 += 32) {   // K over i
    // ---- async global -> LDS stage of wgt[k0..k0+31, mblk..mblk+63] ----
#pragma unroll
    for (int it = 0; it < 4; ++it) {
      int r = sr + it * 8;
      uint64_t g = (uint64_t)(const void*)(Wb + (size_t)(k0 + r) * S2_ + sc);
      unsigned loff = (unsigned)(uintptr_t)&atile[r][sc];  // LDS aperture: addr[31:0]
      asm volatile("global_load_async_to_lds_b128 %0, %1, off"
                   :: "v"(loff), "v"(g) : "memory");
    }
    asm volatile("s_wait_asynccnt 0" ::: "memory");
    __syncthreads();

    // ---- A fragment: row r = j (lane&15), K from LDS rows, exp on read ----
    int rl = wave * 16 + (lane & 15);
    int s = (lane >> 4) * 8;
    v16bf a;
#pragma unroll
    for (int e = 0; e < 8; ++e)
      a[e] = (__bf16)__expf(atile[s + e][rl] - mx);
#pragma unroll
    for (int e = 0; e < 8; ++e)
      a[8 + e] = (__bf16)__expf(atile[16 + s + e][rl] - mx);

    v16bf b0 = ldB_T_bf16(Bb + (size_t) 0 * S1_ + k0, S1_, lane);
    v16bf b1 = ldB_T_bf16(Bb + (size_t)16 * S1_ + k0, S1_, lane);
    v16bf b2 = ldB_T_bf16(Bb + (size_t)32 * S1_ + k0, S1_, lane);
    v16bf b3 = ldB_T_bf16(Bb + (size_t)48 * S1_ + k0, S1_, lane);
    c0 = WMMA_BF16(a, b0, c0);
    c1 = WMMA_BF16(a, b1, c1);
    c2 = WMMA_BF16(a, b2, c2);
    c3 = WMMA_BF16(a, b3, c3);
    __syncthreads();   // protect atile before next stage overwrites
  }

  const float* scale = cinv + (size_t)b * S2_ + m0;
  float* D = out + ((size_t)b * S2_ + m0) * H_ + n0;
  stD_f32_scale(D,      H_, scale, c0, lane);
  stD_f32_scale(D + 16, H_, scale, c1, lane);
  stD_f32_scale(D + 32, H_, scale, c2, lane);
  stD_f32_scale(D + 48, H_, scale, c3, lane);
}

// ---------------- host-side launch ----------------

static inline size_t align256(size_t x) { return (x + 255) & ~(size_t)255; }

extern "C" void kernel_launch(void* const* d_in, const int* in_sizes, int n_in,
                              void* d_out, int out_size, void* d_ws, size_t ws_size,
                              hipStream_t stream) {
  (void)in_sizes; (void)n_in; (void)out_size; (void)ws_size;
  const float* h1   = (const float*)d_in[0];   // (B,S1,H)
  const float* h2   = (const float*)d_in[1];   // (B,S2,H)
  const float* W    = (const float*)d_in[2];   // (H,H)
  const float* bias = (const float*)d_in[3];   // (1,)

  float* out_h1 = (float*)d_out;                               // hidden_1 (B,S2,H)
  float* out_h2 = (float*)d_out + (size_t)B_ * S2_ * H_;       // hidden_2 (B,S1,H)

  char* ws = (char*)d_ws;
  size_t off = 0;
  __bf16* h1w = (__bf16*)(ws + off); off = align256(off + (size_t)B_ * S1_ * H_ * 2);
  __bf16* h2T = (__bf16*)(ws + off); off = align256(off + (size_t)B_ * H_ * S2_ * 2);
  __bf16* h1T = (__bf16*)(ws + off); off = align256(off + (size_t)B_ * H_ * S1_ * 2);
  __bf16* WT  = (__bf16*)(ws + off); off = align256(off + (size_t)H_ * H_ * 2);
  float*  wgt = (float*) (ws + off); off = align256(off + (size_t)B_ * S1_ * S2_ * 4);
  float*  rmax= (float*) (ws + off); off = align256(off + (size_t)B_ * S1_ * 4);
  float*  rinv= (float*) (ws + off); off = align256(off + (size_t)B_ * S1_ * 4);
  float*  cmax= (float*) (ws + off); off = align256(off + (size_t)B_ * S2_ * 4);
  float*  cinv= (float*) (ws + off); off = align256(off + (size_t)B_ * S2_ * 4);

  // transposed bf16 copies (needed so WMMA B-fragment columns are K-contiguous)
  k0_transpose_cvt<<<dim3(H_ / 32, S2_ / 32, B_), 256, 0, stream>>>(h2, h2T, S2_, H_);
  k0_transpose_cvt<<<dim3(H_ / 32, S1_ / 32, B_), 256, 0, stream>>>(h1, h1T, S1_, H_);
  k0_transpose_cvt<<<dim3(H_ / 32, H_ / 32, 1), 256, 0, stream>>>(W, WT, H_, H_);

  // GEMM1: h1w = hidden1 @ W  (bf16 out)
  k1_h1w<<<dim3(H_ / 64, S1_ / 64, B_), 128, 0, stream>>>(h1, WT, h1w);

  // GEMM2: weight = h1w @ hidden2^T + bias (f32)
  k2_weight<<<dim3(S2_ / 64, S1_ / 64, B_), 128, 0, stream>>>(h1w, h2, bias, wgt);

  // softmax statistics
  k3_rowstats<<<dim3(B_ * S1_), 256, 0, stream>>>(wgt, rmax, rinv);
  k4_colstats<<<dim3(S2_ / 256, B_), 256, 0, stream>>>(wgt, cmax, cinv);

  // GEMM3: hidden_2 = softmax_axis2(weight) @ hidden2
  k5_hidden2<<<dim3(H_ / 64, S1_ / 64, B_), 128, 0, stream>>>(wgt, h2T, rmax, rinv, out_h2);

  // GEMM4: hidden_1 = softmax_axis1(weight)^T @ hidden1
  k6_hidden1<<<dim3(H_ / 64, S2_ / 64, B_), 128, 0, stream>>>(wgt, h1T, cmax, cinv, out_h1);
}
